// Conv2dDemodulate_35485019799918
// MI455X (gfx1250) — matrix operand order repaired and checked by reading, not compiled, not run on hardware
//
#include <hip/hip_runtime.h>

// ---------------------------------------------------------------------------
// StyleGAN2 modulated conv (B=4, I=O=64, K=3, H=W=512) as implicit GEMM on
// CDNA5 WMMA bf16. Per sample b:
//   out(O=64, H*W) = A_b(64, 576) x im2col_b(576, H*W)
// A_b = demodulated weights in bf16, packed in WMMA A-fragment order.
// GEMM K index: k = (ky*3+kx)*64 + i  (tap-major, channel-minor) so each
// 32-wide K chunk = one filter tap x one half of the input channels.
// A-fragment staging into LDS uses the Tensor Data Mover when available.
// ---------------------------------------------------------------------------

typedef __attribute__((ext_vector_type(16))) __bf16       v16bf;
typedef __attribute__((ext_vector_type(8)))  float        v8f;
typedef __attribute__((ext_vector_type(4)))  float        v4f;
typedef __attribute__((ext_vector_type(4)))  unsigned int v4u;
typedef __attribute__((ext_vector_type(8)))  int          i32x8;
typedef __attribute__((ext_vector_type(4)))  int          i32x4;

#define BB      4
#define CI      64
#define CO      64
#define KS      3
#define HH      512
#define WW      512
#define KDIM    (9 * CI)          // 576
#define NKC     (KDIM / 32)       // 18 K-chunks
#define NMT     (CO / 16)         // 4 M-tiles
#define TILE_W  128               // pixels per workgroup
#define LCOLS   132               // 130 used + pad
#define LCH     72                // 64 used + pad (16B-aligned stride)
#define FRAG_ELEMS 512            // 32 lanes * 16 bf16 per WMMA A fragment
#define A_PER_B (NMT * NKC * FRAG_ELEMS)   // 36864 bf16 = 73728 B per sample
#define A_QWORDS (A_PER_B * 2 / 8)         // 9216 x 8-byte elements

#if defined(__gfx1250__) && __has_builtin(__builtin_amdgcn_tensor_load_to_lds) && \
    __has_builtin(__builtin_amdgcn_s_wait_tensorcnt)
#define USE_TDM 1
#else
#define USE_TDM 0
#endif

// ---------------------------------------------------------------------------
// Kernel 1: per-(b,o) demodulation factor rsig = 1/sqrt(eps + sum((w*s)^2))
// ---------------------------------------------------------------------------
__global__ void demod_sigma_kernel(const float* __restrict__ style,
                                   const float* __restrict__ weight,
                                   float* __restrict__ rsig) {
    const int t = threadIdx.x;          // 256 threads, 1 block: t = b*64 + o
    const int b = t >> 6;
    const int o = t & 63;
    float sum = 0.f;
    for (int i = 0; i < CI; ++i) {
        const float s = style[b * CI + i];
        const float* wp = weight + (size_t)(o * CI + i) * 9;
#pragma unroll
        for (int k = 0; k < 9; ++k) {
            const float w = wp[k] * s;
            sum += w * w;
        }
    }
    rsig[t] = rsqrtf(1e-8f + sum);
}

// ---------------------------------------------------------------------------
// Kernel 2: pack demodulated bf16 weights directly in WMMA A-fragment layout.
// abuf[(b*NMT + mt)*NKC + kc][lane][e], e -> K per ISA 16-bit A 16x32 layout:
//   K(lane,e) = (e<8 ? e : e+8) + (lane<16 ? 0 : 8)
// ---------------------------------------------------------------------------
__global__ void packA_kernel(const float* __restrict__ style,
                             const float* __restrict__ weight,
                             const float* __restrict__ rsig,
                             __bf16* __restrict__ abuf) {
    const int bid  = blockIdx.x;        // B*NMT*NKC = 288 blocks of 32
    const int lane = threadIdx.x;
    const int b    = bid / (NMT * NKC);
    const int r    = bid % (NMT * NKC);
    const int mt   = r / NKC;
    const int kc   = r % NKC;

    const int o     = mt * 16 + (lane & 15);
    const int tap   = kc >> 1;          // 0..8
    const int ihalf = kc & 1;
    const int ky    = tap / 3;
    const int kx    = tap % 3;
    const int kb    = (lane < 16) ? 0 : 8;
    const float rs  = rsig[b * CO + o];

    __bf16* dst = abuf + (size_t)bid * FRAG_ELEMS + lane * 16;
#pragma unroll
    for (int e = 0; e < 16; ++e) {
        const int kk = ((e < 8) ? e : e + 8) + kb;   // K within chunk, 0..31
        const int i  = ihalf * 32 + kk;              // input channel
        const float v = weight[((size_t)(o * CI + i) * KS + ky) * KS + kx]
                      * style[b * CI + i] * rs;
        dst[e] = (__bf16)v;
    }
}

// ---------------------------------------------------------------------------
// Main kernel: one workgroup (8 wave32) per (b, row y, 128-pixel tile).
//  - stage this sample's packed A fragments (72KB) to LDS via TDM (or copy)
//  - stage input halo (64ch x 3rows x 130cols) to LDS as bf16, [ry][col][ch]
//  - each wave: 16 pixels, loop 18 K-chunks x 4 M-tiles of wmma bf16
// ---------------------------------------------------------------------------
__global__ void __launch_bounds__(256)
conv_wmma_kernel(const float* __restrict__ input,
                 const __bf16* __restrict__ abuf,
                 float* __restrict__ out) {
    __shared__ __bf16 sIn[3 * LCOLS * LCH];   // 57024 B
    __shared__ __bf16 sA[A_PER_B];            // 73728 B

    const int bid = blockIdx.x;               // B*HH*(WW/TILE_W) = 8192
    const int xt  = bid & 3;
    const int y   = (bid >> 2) & (HH - 1);
    const int b   = bid >> 11;
    const int x0  = xt * TILE_W;

    const int tid  = threadIdx.x;
    const int lane = tid & 31;
    const int wid  = tid >> 5;

    // --- stage A fragments for this sample ---
#if USE_TDM
    if (wid == 0) {
        // Tensor DMA: 1-D tile of 9216 x 8-byte elements, global -> LDS.
        const unsigned lds_off = (unsigned)(size_t)(&sA[0]);
        const unsigned long long ga =
            (unsigned long long)(size_t)(abuf + (size_t)b * A_PER_B);
        v4u g0;
        g0.x = 1u;                                      // count=1 (valid user D#)
        g0.y = lds_off;                                 // lds_addr
        g0.z = (unsigned)ga;                            // global_addr[31:0]
        g0.w = (unsigned)(ga >> 32) | (2u << 30);       // global_addr[56:32] | type=2
        i32x8 g1 = {};
        g1[0] = (3 << 16);                              // data_size = 8 bytes
        g1[1] = (int)((A_QWORDS & 0xffff) << 16);       // tensor_dim0[15:0]
        g1[2] = (int)(((A_QWORDS >> 16) & 0xffff) | (1 << 16)); // dim0 hi | dim1=1
        g1[3] = (int)((A_QWORDS & 0xffff) << 16);       // tile_dim0 = 9216
        g1[4] = 1;                                      // tile_dim1 = 1
        g1[5] = (int)A_QWORDS;                          // tensor_dim0_stride lo
        g1[6] = 0;
        g1[7] = 0;
        i32x4 z4 = {};
#if __clang_major__ >= 23
        i32x8 z8 = {};
        __builtin_amdgcn_tensor_load_to_lds(g0, g1, z4, z4, z8, 0);
#else
        __builtin_amdgcn_tensor_load_to_lds(g0, g1, z4, z4, 0);
#endif
        __builtin_amdgcn_s_wait_tensorcnt(0);
    }
#else
    {
        const v4u* src = (const v4u*)(abuf + (size_t)b * A_PER_B);
        v4u* dst = (v4u*)sA;
        for (int idx = tid; idx < A_PER_B / 8; idx += 256)
            dst[idx] = src[idx];
    }
#endif

    // --- stage input halo tile, fp32 -> bf16, layout [ry][col][ch] ---
    // Body cols 1..128 via aligned float4 loads (gx = x0 + 4*lane), halo
    // cols 0 and 129 by lanes 0-1.
    const float* inb = input + (size_t)b * CI * HH * WW;
    for (int task = wid; task < 3 * CI; task += 8) {
        const int ry = task % 3;
        const int ch = task / 3;
        const int gy = y - 1 + ry;
        const bool rowok = (unsigned)gy < (unsigned)HH;
        const float* row = inb + ((size_t)ch * HH + (rowok ? gy : 0)) * WW;

        v4f v = {};
        if (rowok) v = *(const v4f*)(row + x0 + 4 * lane);
        __bf16* dst = &sIn[(ry * LCOLS + 1 + 4 * lane) * LCH + ch];
        dst[0 * LCH] = (__bf16)v.x;
        dst[1 * LCH] = (__bf16)v.y;
        dst[2 * LCH] = (__bf16)v.z;
        dst[3 * LCH] = (__bf16)v.w;

        if (lane < 2) {
            const int c  = lane * 129;            // col 0 or col 129
            const int gx = x0 - 1 + c;            // x0-1 or x0+128
            const float h = (rowok && (unsigned)gx < (unsigned)WW) ? row[gx] : 0.f;
            sIn[(ry * LCOLS + c) * LCH + ch] = (__bf16)h;
        }
    }
    __syncthreads();

    union Frag { v4u u[2]; v16bf v; };

    v8f acc[NMT] = {};

#pragma unroll
    for (int kc = 0; kc < NKC; ++kc) {
        const int tap   = kc >> 1;
        const int ihalf = kc & 1;
        const int ky    = tap / 3;
        const int kx    = tap % 3;

        // B fragment (32x16): lane%16 -> pixel column, K = table order (8h+s, 16+8h+s-8)
        Frag fb;
        {
            const int n     = lane & 15;
            const int c     = wid * 16 + n + kx;               // 0..129
            const int ibase = ihalf * 32 + ((lane < 16) ? 0 : 8);
            const v4u* p = (const v4u*)&sIn[(ky * LCOLS + c) * LCH + ibase];
            fb.u[0] = p[0];
            fb.u[1] = p[2];
        }

#pragma unroll
        for (int mt = 0; mt < NMT; ++mt) {
            Frag fa;
            const v4u* p = (const v4u*)&sA[(mt * NKC + kc) * FRAG_ELEMS + lane * 16];
            fa.u[0] = p[0];
            fa.u[1] = p[1];
            acc[mt] = __builtin_amdgcn_wmma_f32_16x16x32_bf16(
                false, fa.v, false, fb.v, (short)0, acc[mt], false, false);
        }
    }

    // --- epilogue: C layout (lane<16 -> M rows 0..7, lane>=16 -> rows 8..15)
    float* outb = out + (size_t)b * CO * HH * WW;
    const int px   = x0 + wid * 16 + (lane & 15);
    const int mrow = (lane < 16) ? 0 : 8;
#pragma unroll
    for (int mt = 0; mt < NMT; ++mt) {
#pragma unroll
        for (int r2 = 0; r2 < 8; ++r2) {
            const int o = mt * 16 + mrow + r2;
            outb[((size_t)o * HH + y) * WW + px] = acc[mt][r2];
        }
    }
}

// ---------------------------------------------------------------------------
extern "C" void kernel_launch(void* const* d_in, const int* in_sizes, int n_in,
                              void* d_out, int out_size, void* d_ws, size_t ws_size,
                              hipStream_t stream) {
    (void)in_sizes; (void)n_in; (void)out_size; (void)ws_size;

    const float* input  = (const float*)d_in[0];   // (B, I, H, W) f32
    const float* style  = (const float*)d_in[1];   // (B, I)       f32
    const float* weight = (const float*)d_in[2];   // (O, I, K, K) f32
    float* out = (float*)d_out;                    // (B, O, H, W) f32

    // workspace: [A fragments bf16: 294912 B][rsig f32: 1024 B]
    __bf16* abuf = (__bf16*)d_ws;
    float*  rsig = (float*)((char*)d_ws + (size_t)BB * A_PER_B * sizeof(__bf16));

    demod_sigma_kernel<<<1, 256, 0, stream>>>(style, weight, rsig);
    packA_kernel<<<BB * NMT * NKC, 32, 0, stream>>>(style, weight, rsig, abuf);
    conv_wmma_kernel<<<BB * HH * (WW / TILE_W), 256, 0, stream>>>(input, abuf, out);
}
